// GRU_51608327029091
// MI455X (gfx1250) — compile-verified, matrix operand order checked by reading, not run on hardware
//
#include <hip/hip_runtime.h>
#include <hip/hip_bf16.h>

typedef __attribute__((ext_vector_type(16))) _Float16 v16h;
typedef __attribute__((ext_vector_type(8)))  _Float16 v8h;
typedef __attribute__((ext_vector_type(8)))  float    v8f;

constexpr int T = 35, B = 128, E = 200, H = 512, L = 2, V = 10000;
constexpr int KX0 = 224;  // E padded to multiple of 32

// ---------------- fragment load helpers ----------------
// 16-bit A/B fragment per ISA layout: lane r=lane&15 owns row (A: M=r, B: N=r),
// lanes 16..31 hold the +8 k-halves. Element e<8 -> k0+hsel*8+e ; e>=8 -> k0+16+hsel*8+(e-8).
__device__ __forceinline__ v16h load_frag(const _Float16* __restrict__ p) {
  v8h lo = *(const v8h*)(p);
  v8h hi = *(const v8h*)(p + 16);
  v16h r;
#pragma unroll
  for (int i = 0; i < 8; ++i) { r[i] = lo[i]; r[i + 8] = hi[i]; }
  return r;
}

// Software-pipelined register-tile GEMM: C[MT*16, NT*16] += A[m0.., K] * W[n0.., K]^T
// Prefetches the k+32 fragments before issuing the WMMAs for chunk k so loads
// overlap matrix ops (avoids s_wait_loadcnt 0 on the critical path).
template <int MT, int NT>
__device__ __forceinline__ void gemm_tile(const _Float16* __restrict__ A, int lda,
                                          const _Float16* __restrict__ W, int ldw,
                                          int m0, int n0, int K, v8f* acc) {
  const int lane = threadIdx.x & 31;
  const int r    = lane & 15;
  const int hsel = (lane >> 4) << 3;  // 0 or 8

  const _Float16* ap[MT];
  const _Float16* wp[NT];
#pragma unroll
  for (int mt = 0; mt < MT; ++mt) ap[mt] = A + (size_t)(m0 + mt * 16 + r) * lda + hsel;
#pragma unroll
  for (int nt = 0; nt < NT; ++nt) wp[nt] = W + (size_t)(n0 + nt * 16 + r) * ldw + hsel;

  v16h a[MT], b[NT];
#pragma unroll
  for (int mt = 0; mt < MT; ++mt) a[mt] = load_frag(ap[mt]);
#pragma unroll
  for (int nt = 0; nt < NT; ++nt) b[nt] = load_frag(wp[nt]);

#pragma unroll 2
  for (int k0 = 32; k0 < K; k0 += 32) {
    v16h a2[MT], b2[NT];
#pragma unroll
    for (int mt = 0; mt < MT; ++mt) a2[mt] = load_frag(ap[mt] + k0);
#pragma unroll
    for (int nt = 0; nt < NT; ++nt) b2[nt] = load_frag(wp[nt] + k0);
#pragma unroll
    for (int nt = 0; nt < NT; ++nt)
#pragma unroll
      for (int mt = 0; mt < MT; ++mt)
        acc[mt * NT + nt] = __builtin_amdgcn_wmma_f32_16x16x32_f16(
            false, a[mt], false, b[nt], (short)0, acc[mt * NT + nt], false, false);
#pragma unroll
    for (int mt = 0; mt < MT; ++mt) a[mt] = a2[mt];
#pragma unroll
    for (int nt = 0; nt < NT; ++nt) b[nt] = b2[nt];
  }
#pragma unroll
  for (int nt = 0; nt < NT; ++nt)
#pragma unroll
    for (int mt = 0; mt < MT; ++mt)
      acc[mt * NT + nt] = __builtin_amdgcn_wmma_f32_16x16x32_f16(
          false, a[mt], false, b[nt], (short)0, acc[mt * NT + nt], false, false);
}

// ---------------- prep kernels ----------------
__global__ void cvt_f16_kernel(const float* __restrict__ src, _Float16* __restrict__ dst, int n) {
  int i = blockIdx.x * blockDim.x + threadIdx.x;
  if (i < n) dst[i] = (_Float16)src[i];
}

__global__ void cvt_pad_kernel(const float* __restrict__ src, _Float16* __restrict__ dst,
                               int rows, int K, int Kp) {
  int i = blockIdx.x * blockDim.x + threadIdx.x;
  if (i < rows * Kp) {
    int r = i / Kp, c = i % Kp;
    dst[i] = (c < K) ? (_Float16)src[(size_t)r * K + c] : (_Float16)0.0f;
  }
}

__global__ void embed_kernel(const int* __restrict__ inputs, const float* __restrict__ emb,
                             _Float16* __restrict__ x16, int n) {
  int i = blockIdx.x * blockDim.x + threadIdx.x;
  if (i < n) {
    int c = i % KX0, tb = i / KX0;
    int tok = inputs[tb];
    x16[i] = (c < E) ? (_Float16)emb[(size_t)tok * E + c] : (_Float16)0.0f;
  }
}

__global__ void hidden_init_kernel(const float* __restrict__ hin, float* __restrict__ hout,
                                   _Float16* __restrict__ h0, _Float16* __restrict__ h1) {
  int i = blockIdx.x * blockDim.x + threadIdx.x;
  if (i < L * B * H) {
    float v = hin[i];
    hout[i] = v;
    if (i < B * H) h0[i] = (_Float16)v; else h1[i - B * H] = (_Float16)v;
  }
}

// ---------------- GRU step kernels ----------------
// Phase 1: gates r,f for one layer; also stashes raw gx for the candidate gate.
// C = data16[B,Kx] @ Wx16[3H,Kx]^T ; cols < 2H additionally += h16[B,H] @ Wh16[rows 0..2H)[.,H]^T
__global__ __launch_bounds__(256) void gates_rf_kernel(
    const _Float16* __restrict__ data16, int lda, int Kx,
    const _Float16* __restrict__ h16,
    const _Float16* __restrict__ Wx16,   // [3H, Kx]
    const _Float16* __restrict__ Wh16,   // layer base, rows [3H, H]
    const float* __restrict__ bh,        // layer base [3,H]
    const float* __restrict__ hidden,    // fp32 [B,H]
    float* __restrict__ Fg, _Float16* __restrict__ RH, float* __restrict__ gx2) {
  const int NT = 4;
  const int nW = (3 * H) / (16 * NT);              // 24 wave tiles along N
  int wave = blockIdx.x * (blockDim.x >> 5) + (threadIdx.x >> 5);
  int mt = wave / nW, ntw = wave % nW;
  int m0 = mt * 16, n0 = ntw * 16 * NT;

  v8f acc[NT] = {};
  gemm_tile<1, NT>(data16, lda, Wx16, Kx, m0, n0, Kx, acc);
  if (n0 < 2 * H) gemm_tile<1, NT>(h16, H, Wh16, H, m0, n0, H, acc);

  const int lane = threadIdx.x & 31;
  const int rc = lane & 15, hs = lane >> 4;
#pragma unroll
  for (int nt = 0; nt < NT; ++nt) {
    int n = n0 + nt * 16 + rc;
    int gate = n >> 9;       // /H
    int j = n & (H - 1);
#pragma unroll
    for (int vr = 0; vr < 8; ++vr) {
      int m = m0 + vr + (hs << 3);
      float v = acc[nt][vr];
      if (gate == 2) {
        gx2[m * H + j] = v;
      } else {
        v += bh[gate * H + j];
        float s = 1.0f / (1.0f + __expf(-v));
        if (gate == 0) RH[m * H + j] = (_Float16)(s * hidden[m * H + j]);
        else           Fg[m * H + j] = s;
      }
    }
  }
}

// Phase 2: candidate + state update for one layer.
__global__ __launch_bounds__(256) void cand_kernel(
    const _Float16* __restrict__ RH,
    const _Float16* __restrict__ Wh2,    // rows [H, H] (gate-2 block)
    const float* __restrict__ bh2,       // [H]
    const float* __restrict__ gx2, const float* __restrict__ Fg,
    float* __restrict__ hidden,          // fp32 [B,H], read+write
    _Float16* __restrict__ h16out, _Float16* __restrict__ htop /*nullable*/) {
  const int NT = 4;
  const int nW = H / (16 * NT);          // 8
  int wave = blockIdx.x * (blockDim.x >> 5) + (threadIdx.x >> 5);
  int mt = wave / nW, ntw = wave % nW;
  int m0 = mt * 16, n0 = ntw * 16 * NT;

  v8f acc[NT] = {};
  gemm_tile<1, NT>(RH, H, Wh2, H, m0, n0, H, acc);

  const int lane = threadIdx.x & 31;
  const int rc = lane & 15, hs = lane >> 4;
#pragma unroll
  for (int nt = 0; nt < NT; ++nt) {
    int n = n0 + nt * 16 + rc;
#pragma unroll
    for (int vr = 0; vr < 8; ++vr) {
      int m = m0 + vr + (hs << 3);
      float c  = tanhf(acc[nt][vr] + gx2[m * H + n] + bh2[n]);
      float hp = hidden[m * H + n];
      float f  = Fg[m * H + n];
      float hn = (1.0f - f) * hp + f * c;
      hidden[m * H + n] = hn;
      h16out[m * H + n] = (_Float16)hn;
      if (htop) htop[m * H + n] = (_Float16)hn;
    }
  }
}

// Final projection: [T*B, H] x [V, H]^T + bout -> [T*B, V]
// 32x80 per wave (MT=2 x NT=5): B fragments reused 2x, A fragments 5x.
__global__ __launch_bounds__(256) void logits_kernel(
    const _Float16* __restrict__ A,       // htop16 [T*B, H]
    const _Float16* __restrict__ Wout16,  // [V, H]
    const float* __restrict__ bout, float* __restrict__ out) {
  const int MT = 2, NT = 5;
  const int nW = V / (16 * NT);                    // 125
  const int totalWaves = ((T * B) / (16 * MT)) * nW;  // 140*125 = 17500
  int wave = blockIdx.x * (blockDim.x >> 5) + (threadIdx.x >> 5);
  if (wave >= totalWaves) return;
  int mt = wave / nW, ntw = wave % nW;
  int m0 = mt * 16 * MT, n0 = ntw * 16 * NT;

  v8f acc[MT * NT] = {};
  gemm_tile<MT, NT>(A, H, Wout16, H, m0, n0, H, acc);

  const int lane = threadIdx.x & 31;
  const int rc = lane & 15, hs = lane >> 4;
#pragma unroll
  for (int nt = 0; nt < NT; ++nt) {
    int n = n0 + nt * 16 + rc;
    float bb = bout[n];
#pragma unroll
    for (int mt2 = 0; mt2 < MT; ++mt2) {
#pragma unroll
      for (int vr = 0; vr < 8; ++vr) {
        int m = m0 + mt2 * 16 + vr + (hs << 3);
        out[(size_t)m * V + n] = acc[mt2 * NT + nt][vr] + bb;
      }
    }
  }
}

// ---------------- host orchestration ----------------
extern "C" void kernel_launch(void* const* d_in, const int* in_sizes, int n_in,
                              void* d_out, int out_size, void* d_ws, size_t ws_size,
                              hipStream_t stream) {
  (void)in_sizes; (void)n_in; (void)out_size; (void)ws_size;
  const int*   inputs    = (const int*)d_in[0];
  const float* hidden_in = (const float*)d_in[1];
  const float* emb       = (const float*)d_in[2];
  const float* Wx0       = (const float*)d_in[3];
  const float* Wx1       = (const float*)d_in[4];
  const float* Wh        = (const float*)d_in[5];
  const float* bh        = (const float*)d_in[6];
  const float* Wout      = (const float*)d_in[7];
  const float* bout      = (const float*)d_in[8];

  float* out  = (float*)d_out;                  // logits [T,B,V]
  float* hout = out + (size_t)T * B * V;        // live hidden / h_final [L,B,H]

  // workspace carve-up (256B aligned slabs)
  char* wsb = (char*)d_ws;
  size_t off = 0;
  auto take = [&](size_t bytes) -> char* {
    off = (off + 255) & ~(size_t)255;
    char* p = wsb + off;
    off += bytes;
    return p;
  };
  _Float16* x16    = (_Float16*)take((size_t)T * B * KX0 * 2);
  _Float16* wx0_16 = (_Float16*)take((size_t)3 * H * KX0 * 2);
  _Float16* wx1_16 = (_Float16*)take((size_t)3 * H * H * 2);
  _Float16* wh16   = (_Float16*)take((size_t)L * 3 * H * H * 2);
  _Float16* wout16 = (_Float16*)take((size_t)V * H * 2);
  _Float16* htop16 = (_Float16*)take((size_t)T * B * H * 2);
  _Float16* h16_0  = (_Float16*)take((size_t)B * H * 2);
  _Float16* h16_1  = (_Float16*)take((size_t)B * H * 2);
  _Float16* RH     = (_Float16*)take((size_t)B * H * 2);
  float*    Fg     = (float*)take((size_t)B * H * 4);
  float*    gx2    = (float*)take((size_t)B * H * 4);

  const int TPB = 256;
  auto cdiv = [](int a, int b) { return (a + b - 1) / b; };

  // one-time (per launch) weight / input conversion
  cvt_pad_kernel<<<cdiv(3 * H * KX0, TPB), TPB, 0, stream>>>(Wx0, wx0_16, 3 * H, E, KX0);
  cvt_f16_kernel<<<cdiv(3 * H * H, TPB), TPB, 0, stream>>>(Wx1, wx1_16, 3 * H * H);
  cvt_f16_kernel<<<cdiv(L * 3 * H * H, TPB), TPB, 0, stream>>>(Wh, wh16, L * 3 * H * H);
  cvt_f16_kernel<<<cdiv(V * H, TPB), TPB, 0, stream>>>(Wout, wout16, V * H);
  embed_kernel<<<cdiv(T * B * KX0, TPB), TPB, 0, stream>>>(inputs, emb, x16, T * B * KX0);
  hidden_init_kernel<<<cdiv(L * B * H, TPB), TPB, 0, stream>>>(hidden_in, hout, h16_0, h16_1);

  const int rfBlocks   = ((B / 16) * ((3 * H) / 64)) / 8;  // 24
  const int candBlocks = ((B / 16) * (H / 64)) / 8;        // 8

  for (int t = 0; t < T; ++t) {
    // layer 0
    gates_rf_kernel<<<rfBlocks, TPB, 0, stream>>>(
        x16 + (size_t)t * B * KX0, KX0, KX0, h16_0,
        wx0_16, wh16, bh, hout, Fg, RH, gx2);
    cand_kernel<<<candBlocks, TPB, 0, stream>>>(
        RH, wh16 + (size_t)2 * H * H, bh + 2 * H, gx2, Fg,
        hout, h16_0, (_Float16*)nullptr);
    // layer 1 (data = h0_new in f16)
    gates_rf_kernel<<<rfBlocks, TPB, 0, stream>>>(
        h16_0, H, H, h16_1,
        wx1_16, wh16 + (size_t)3 * H * H, bh + 3 * H, hout + (size_t)B * H,
        Fg, RH, gx2);
    cand_kernel<<<candBlocks, TPB, 0, stream>>>(
        RH, wh16 + (size_t)(3 * H + 2 * H) * H, bh + 3 * H + 2 * H, gx2, Fg,
        hout + (size_t)B * H, h16_1, htop16 + (size_t)t * B * H);
  }

  // big independent GEMM: logits (32x80 wave tiles)
  const int totalWaves = ((T * B) / 32) * (V / 80);   // 17500
  logits_kernel<<<cdiv(totalWaves, 8), TPB, 0, stream>>>(htop16, wout16, bout, out);
}
